// LatentPerturber_11175504904888
// MI455X (gfx1250) — compile-verified, matrix operand order checked by reading, not run on hardware
//
#include <hip/hip_runtime.h>
#include <hip/hip_bf16.h>
#include <stddef.h>

// ---------------------------------------------------------------------------
// Types for CDNA5 WMMA (wave32, 16x16x32 bf16 -> f32)
// ---------------------------------------------------------------------------
typedef __attribute__((ext_vector_type(16))) __bf16 v16bf;
typedef __attribute__((ext_vector_type(8)))  __bf16 v8bf;
typedef __attribute__((ext_vector_type(8)))  float  v8f;

#define N_ROWS   8192
#define DIMS     512
#define HID      1024
#define KSEL     4
#define FBIG     3.402823466e38f
#define LEAKK    0.01f
#define HLOG2PI  0.91893853320467274f

// d_out layout (floats): x_out | entropy | rows | subsets
#define OUT_XOUT 0
#define OUT_ENT  (N_ROWS * DIMS)                 // 4194304
#define OUT_ROWS (OUT_ENT + 1)                   // 4194305
#define OUT_SUBS (OUT_ROWS + N_ROWS * KSEL)      // 4227073

// workspace layout (bytes)
#define WS_XB      0                             // 8192*512*2   = 8388608
#define WS_WB1     8388608                       // 1024*1024*2  = 2097152
#define WS_WB2     10485760                      // 2097152
#define WS_SQ      12582912                      // 8192*4       = 32768
#define WS_SUBI    12615680                      // 8192*4*4     = 131072
#define WS_H0      12746752                      // 8192*1024*2  = 16777216
#define WS_H1      29523968                      // 16777216
#define WS_H2      46301184                      // 8192*1024*4  = 33554432
#define WS_ROWSUM  79855616                      // 32768
#define WS_CANDD   79888384                      // 8192*4*8*4   = 1048576
#define WS_CANDI   80936960                      // 1048576

#define WMMA_BF16(A, B, C) \
  __builtin_amdgcn_wmma_f32_16x16x32_bf16(false, (A), false, (B), (short)0, (C), false, false)

// ---------------------------------------------------------------------------
// Fragment loader: lane l holds row (base + (l&15)), K block
// [k0 + koff .. +7] and [k0 + 16 + koff .. +7] with koff = 8*(l>=16).
// Two 16B (b128) loads per lane. `stride` = K length of the matrix.
// ---------------------------------------------------------------------------
__device__ __forceinline__ v16bf load_frag(const __bf16* __restrict__ base,
                                           int stride, int row0, int k0, int lane) {
  const int r    = row0 + (lane & 15);
  const int koff = k0 + ((lane >> 4) << 3);
  const __bf16* p = base + (size_t)r * (size_t)stride + koff;
  v8bf lo = *(const v8bf*)(p);
  v8bf hi = *(const v8bf*)(p + 16);
  return __builtin_shufflevector(lo, hi, 0, 1, 2, 3, 4, 5, 6, 7,
                                         8, 9, 10, 11, 12, 13, 14, 15);
}

// Strict '<' insert: within one scanning lane, discovery order == ascending
// column index, so equal values keep the earlier (lower) index first.
__device__ __forceinline__ void topk_insert(float dist, int idx,
                                            float& d0, float& d1, float& d2, float& d3,
                                            int& i0, int& i1, int& i2, int& i3) {
  if (dist < d3) {
    if (dist < d2) {
      d3 = d2; i3 = i2;
      if (dist < d1) {
        d2 = d1; i2 = i1;
        if (dist < d0) { d1 = d0; i1 = i0; d0 = dist; i0 = idx; }
        else           { d1 = dist; i1 = idx; }
      } else { d2 = dist; i2 = idx; }
    } else { d3 = dist; i3 = idx; }
  }
}

// (dist, idx)-lexicographic insert for merging candidate lists: reproduces
// jax.lax.top_k's lower-index tie-break regardless of list interleaving.
__device__ __forceinline__ void topk_insert_lex(float dist, int idx,
                                                float& d0, float& d1, float& d2, float& d3,
                                                int& i0, int& i1, int& i2, int& i3) {
  if ((dist < d3) || (dist == d3 && idx < i3)) {
    if ((dist < d2) || (dist == d2 && idx < i2)) {
      d3 = d2; i3 = i2;
      if ((dist < d1) || (dist == d1 && idx < i1)) {
        d2 = d1; i2 = i1;
        if ((dist < d0) || (dist == d0 && idx < i0)) {
          d1 = d0; i1 = i0; d0 = dist; i0 = idx;
        } else { d1 = dist; i1 = idx; }
      } else { d2 = dist; i2 = idx; }
    } else { d3 = dist; i3 = idx; }
  }
}

// ---------------------------------------------------------------------------
// K1: convert x -> bf16 and compute per-row squared norms. One wave per row.
// ---------------------------------------------------------------------------
__global__ __launch_bounds__(32) void k_prep_x(const float* __restrict__ x,
                                               __bf16* __restrict__ xb,
                                               float* __restrict__ sq) {
  const int row  = blockIdx.x;
  const int lane = threadIdx.x;
  float s = 0.0f;
  for (int j = lane; j < DIMS; j += 32) {
    const float v = x[(size_t)row * DIMS + j];
    s += v * v;
    xb[(size_t)row * DIMS + j] = (__bf16)v;
  }
  for (int off = 16; off > 0; off >>= 1) s += __shfl_down(s, off, 32);
  if (lane == 0) sq[row] = s;
}

// ---------------------------------------------------------------------------
// K2: convert W1/W2 -> bf16 (flat).
// ---------------------------------------------------------------------------
__global__ __launch_bounds__(256) void k_conv_w(const float* __restrict__ W1,
                                                const float* __restrict__ W2,
                                                __bf16* __restrict__ wb1,
                                                __bf16* __restrict__ wb2) {
  const size_t i = (size_t)blockIdx.x * 256 + threadIdx.x;
  const size_t M = (size_t)HID * HID;
  if (i < M)            wb1[i]     = (__bf16)W1[i];
  else if (i < 2 * M)   wb2[i - M] = (__bf16)W2[i - M];
}

// ---------------------------------------------------------------------------
// K3: fused Gram + distance + per-quarter top-4 half-lists.
// grid = (4 column-quarters, 512 row-strips), 1 wave per block.
// 16 rows x 32 cols per iteration, 2 accumulator chains, depth-2 B prefetch
// (8 loads in flight). Epilogue: all 32 lanes scan (row = l&15, col-half =
// l>=16), 16 inserts/lane/iteration; 8 candidates per row emitted at the end.
// ---------------------------------------------------------------------------
__global__ __launch_bounds__(32) void k_gram_topk(const __bf16* __restrict__ xb,
                                                  const float* __restrict__ sq,
                                                  float* __restrict__ cand_d,
                                                  int* __restrict__ cand_i) {
  const int lane    = threadIdx.x;
  const int quarter = blockIdx.x;
  const int rowBase = blockIdx.y * 16;
  __shared__ float ld[16][33];

  // Hoist the A strip (16 rows x 512 K) into registers: 16 fragments.
  v16bf a[16];
#pragma unroll
  for (int kk = 0; kk < 16; ++kk)
    a[kk] = load_frag(xb, DIMS, rowBase, kk * 32, lane);

  const int mb = (lane >> 4) << 3;
  float sqr[8];
#pragma unroll
  for (int v = 0; v < 8; ++v) sqr[v] = sq[rowBase + mb + v];

  const int r    = lane & 15;       // scanned row within strip
  const int half = lane >> 4;       // scanned column half (0: 0-15, 1: 16-31)
  float d0 = FBIG, d1 = FBIG, d2 = FBIG, d3 = FBIG;
  int   i0 = 0,    i1 = 0,    i2 = 0,    i3 = 0;

  for (int it = 0; it < 64; ++it) {              // 64 x 32 cols = 2048 cols
    const int colBase = (quarter * 64 + it) * 32;
    v8f acc0 = {}, acc1 = {};
    // depth-2 rotating prefetch: cur (c), next (n), incoming (p)
    v16bf c0 = load_frag(xb, DIMS, colBase,      0,  lane);
    v16bf c1 = load_frag(xb, DIMS, colBase + 16, 0,  lane);
    v16bf n0 = load_frag(xb, DIMS, colBase,      32, lane);
    v16bf n1 = load_frag(xb, DIMS, colBase + 16, 32, lane);
#pragma unroll
    for (int kk = 0; kk < 16; ++kk) {
      v16bf p0, p1;
      if (kk < 14) {
        p0 = load_frag(xb, DIMS, colBase,      (kk + 2) * 32, lane);
        p1 = load_frag(xb, DIMS, colBase + 16, (kk + 2) * 32, lane);
      }
      acc0 = WMMA_BF16(a[kk], c0, acc0);
      acc1 = WMMA_BF16(a[kk], c1, acc1);
      c0 = n0; c1 = n1;
      if (kk < 14) { n0 = p0; n1 = p1; }
    }
    const float sqc0 = sq[colBase + (lane & 15)];
    const float sqc1 = sq[colBase + 16 + (lane & 15)];
#pragma unroll
    for (int v = 0; v < 8; ++v) {
      ld[mb + v][lane & 15]        = sqr[v] + sqc0 - 2.0f * acc0[v];
      ld[mb + v][16 + (lane & 15)] = sqr[v] + sqc1 - 2.0f * acc1[v];
    }
    // Single-wave block: DS ops are in-order per wave; fence for visibility.
    asm volatile("s_wait_dscnt 0x0" ::: "memory");
#pragma unroll
    for (int nn = 0; nn < 16; ++nn) {
      const int cc = half * 16 + nn;
      topk_insert(ld[r][cc], colBase + cc, d0, d1, d2, d3, i0, i1, i2, i3);
    }
    asm volatile("" ::: "memory");
  }

  // Emit this lane's half-list: 8 candidates per row per quarter.
  const int row = rowBase + r;
  const size_t b = (((size_t)row * 4 + quarter) * 2 + half) * 4;
  cand_d[b + 0] = d0; cand_d[b + 1] = d1; cand_d[b + 2] = d2; cand_d[b + 3] = d3;
  cand_i[b + 0] = i0; cand_i[b + 1] = i1; cand_i[b + 2] = i2; cand_i[b + 3] = i3;
}

// ---------------------------------------------------------------------------
// K4: merge the 32 candidates per row -> final top-4 (lexicographic insert).
// ---------------------------------------------------------------------------
__global__ __launch_bounds__(128) void k_topk_merge(const float* __restrict__ cand_d,
                                                    const int* __restrict__ cand_i,
                                                    int* __restrict__ subsets,
                                                    float* __restrict__ out) {
  const int row = blockIdx.x * 128 + threadIdx.x;
  if (row >= N_ROWS) return;
  float d0 = FBIG, d1 = FBIG, d2 = FBIG, d3 = FBIG;
  int   i0 = 0x7fffffff, i1 = 0x7fffffff, i2 = 0x7fffffff, i3 = 0x7fffffff;
#pragma unroll
  for (int q = 0; q < 32; ++q) {
    const size_t o = (size_t)row * 32 + q;
    topk_insert_lex(cand_d[o], cand_i[o], d0, d1, d2, d3, i0, i1, i2, i3);
  }
  subsets[row * 4 + 0] = i0; subsets[row * 4 + 1] = i1;
  subsets[row * 4 + 2] = i2; subsets[row * 4 + 3] = i3;
  float* rows_out = out + OUT_ROWS;
  float* subs_out = out + OUT_SUBS;
  const float fr = (float)row;
  rows_out[row * 4 + 0] = fr; rows_out[row * 4 + 1] = fr;
  rows_out[row * 4 + 2] = fr; rows_out[row * 4 + 3] = fr;
  subs_out[row * 4 + 0] = (float)i0; subs_out[row * 4 + 1] = (float)i1;
  subs_out[row * 4 + 2] = (float)i2; subs_out[row * 4 + 3] = (float)i3;
}

// ---------------------------------------------------------------------------
// K5: gather the 4 neighbor rows, mean/max pool in fp32, emit h0 in bf16.
// ---------------------------------------------------------------------------
__global__ __launch_bounds__(128) void k_pool(const float* __restrict__ x,
                                              const int* __restrict__ subsets,
                                              __bf16* __restrict__ h0) {
  const int row = blockIdx.x;
  const int t   = threadIdx.x;
  const int j0 = subsets[row * 4 + 0], j1 = subsets[row * 4 + 1];
  const int j2 = subsets[row * 4 + 2], j3 = subsets[row * 4 + 3];
  for (int j = t; j < DIMS; j += 128) {
    const float a = x[(size_t)j0 * DIMS + j];
    const float b = x[(size_t)j1 * DIMS + j];
    const float c = x[(size_t)j2 * DIMS + j];
    const float d = x[(size_t)j3 * DIMS + j];
    const float mean = (a + b + c + d) * 0.25f;
    const float mx   = fmaxf(fmaxf(a, b), fmaxf(c, d));
    h0[(size_t)row * HID + j]        = (__bf16)mean;
    h0[(size_t)row * HID + DIMS + j] = (__bf16)mx;
  }
}

// ---------------------------------------------------------------------------
// 32x32 tile GEMM core with depth-2 prefetch: C[2][2] = A @ W^T over K=HID.
// 4 independent WMMA chains, each fragment used twice, 16 loads in flight.
// ---------------------------------------------------------------------------
__device__ __forceinline__ void gemm_tile_2x2(const __bf16* __restrict__ A,
                                              const __bf16* __restrict__ W,
                                              int m0, int n0, int lane,
                                              v8f acc[4]) {
  const v8f z = {};
  acc[0] = z; acc[1] = z; acc[2] = z; acc[3] = z;
  v16bf a0c = load_frag(A, HID, m0,      0,  lane);
  v16bf a1c = load_frag(A, HID, m0 + 16, 0,  lane);
  v16bf b0c = load_frag(W, HID, n0,      0,  lane);
  v16bf b1c = load_frag(W, HID, n0 + 16, 0,  lane);
  v16bf a0n = load_frag(A, HID, m0,      32, lane);
  v16bf a1n = load_frag(A, HID, m0 + 16, 32, lane);
  v16bf b0n = load_frag(W, HID, n0,      32, lane);
  v16bf b1n = load_frag(W, HID, n0 + 16, 32, lane);
#pragma unroll 4
  for (int k0 = 0; k0 < HID; k0 += 32) {
    v16bf a0p, a1p, b0p, b1p;
    const bool more2 = (k0 + 64) < HID;
    if (more2) {
      a0p = load_frag(A, HID, m0,      k0 + 64, lane);
      a1p = load_frag(A, HID, m0 + 16, k0 + 64, lane);
      b0p = load_frag(W, HID, n0,      k0 + 64, lane);
      b1p = load_frag(W, HID, n0 + 16, k0 + 64, lane);
    }
    acc[0] = WMMA_BF16(a0c, b0c, acc[0]);
    acc[1] = WMMA_BF16(a0c, b1c, acc[1]);
    acc[2] = WMMA_BF16(a1c, b0c, acc[2]);
    acc[3] = WMMA_BF16(a1c, b1c, acc[3]);
    a0c = a0n; a1c = a1n; b0c = b0n; b1c = b1n;
    if (more2) { a0n = a0p; a1n = a1p; b0n = b0p; b1n = b1p; }
  }
}

// ---------------------------------------------------------------------------
// K6: GEMM1: C = leaky_relu(A @ W^T + bias) in bf16. 1 wave per 32x32 tile.
// ---------------------------------------------------------------------------
__global__ __launch_bounds__(32) void k_gemm_lrelu(const __bf16* __restrict__ A,
                                                   const __bf16* __restrict__ W,
                                                   const float* __restrict__ bias,
                                                   __bf16* __restrict__ C) {
  const int lane = threadIdx.x;
  const int n0 = blockIdx.x * 32;
  const int m0 = blockIdx.y * 32;
  v8f acc[4];
  gemm_tile_2x2(A, W, m0, n0, lane, acc);
  const int mb = (lane >> 4) << 3;
#pragma unroll
  for (int q = 0; q < 4; ++q) {
    const int mi = q >> 1, ni = q & 1;
    const int col = n0 + ni * 16 + (lane & 15);
    const float bv = bias[col];
#pragma unroll
    for (int v = 0; v < 8; ++v) {
      float h = acc[q][v] + bv;
      h = (h >= 0.0f) ? h : LEAKK * h;
      C[(size_t)(m0 + mi * 16 + mb + v) * HID + col] = (__bf16)h;
    }
  }
}

// ---------------------------------------------------------------------------
// K7: GEMM2: C = A @ W^T + bias, fp32 output (mu/log-var halves split later).
// ---------------------------------------------------------------------------
__global__ __launch_bounds__(32) void k_gemm_f32(const __bf16* __restrict__ A,
                                                 const __bf16* __restrict__ W,
                                                 const float* __restrict__ bias,
                                                 float* __restrict__ C) {
  const int lane = threadIdx.x;
  const int n0 = blockIdx.x * 32;
  const int m0 = blockIdx.y * 32;
  v8f acc[4];
  gemm_tile_2x2(A, W, m0, n0, lane, acc);
  const int mb = (lane >> 4) << 3;
#pragma unroll
  for (int q = 0; q < 4; ++q) {
    const int mi = q >> 1, ni = q & 1;
    const int col = n0 + ni * 16 + (lane & 15);
    const float bv = bias[col];
#pragma unroll
    for (int v = 0; v < 8; ++v)
      C[(size_t)(m0 + mi * 16 + mb + v) * HID + col] = acc[q][v] + bv;
  }
}

// ---------------------------------------------------------------------------
// K8: reparameterize: x_out = x + mu + exp(0.5*h_sigma)*eps; per-row h_sigma sum.
// ---------------------------------------------------------------------------
__global__ __launch_bounds__(128) void k_final(const float* __restrict__ x,
                                               const float* __restrict__ eps,
                                               const float* __restrict__ h2,
                                               float* __restrict__ xout,
                                               float* __restrict__ rowsum) {
  const int row = blockIdx.x;
  const int t   = threadIdx.x;
  __shared__ float red[128];
  float s = 0.0f;
  for (int j = t; j < DIMS; j += 128) {
    const float mu = h2[(size_t)row * HID + j];
    const float hs = h2[(size_t)row * HID + DIMS + j];
    const float sg = expf(0.5f * hs);
    xout[(size_t)row * DIMS + j] =
        x[(size_t)row * DIMS + j] + mu + sg * eps[(size_t)row * DIMS + j];
    s += hs;
  }
  red[t] = s;
  __syncthreads();
  for (int off = 64; off > 0; off >>= 1) {
    if (t < off) red[t] += red[t + off];
    __syncthreads();
  }
  if (t == 0) rowsum[row] = red[0];
}

// ---------------------------------------------------------------------------
// K9: deterministic entropy reduction.
// entropy = 0.5 + 0.5*log(2pi) + mean(0.5*h_sigma)
// ---------------------------------------------------------------------------
__global__ __launch_bounds__(256) void k_entropy(const float* __restrict__ rowsum,
                                                 float* __restrict__ out_ent) {
  __shared__ float red[256];
  const int t = threadIdx.x;
  float s = 0.0f;
  for (int i = t; i < N_ROWS; i += 256) s += rowsum[i];
  red[t] = s;
  __syncthreads();
  for (int off = 128; off > 0; off >>= 1) {
    if (t < off) red[t] += red[t + off];
    __syncthreads();
  }
  if (t == 0)
    *out_ent = 0.5f + HLOG2PI +
               0.5f * (red[0] / ((float)N_ROWS * (float)DIMS));
}

// ---------------------------------------------------------------------------
// Host-side launcher
// ---------------------------------------------------------------------------
extern "C" void kernel_launch(void* const* d_in, const int* in_sizes, int n_in,
                              void* d_out, int out_size, void* d_ws, size_t ws_size,
                              hipStream_t stream) {
  (void)in_sizes; (void)n_in; (void)out_size; (void)ws_size;
  const float* x   = (const float*)d_in[0];
  const float* W1  = (const float*)d_in[1];
  const float* b1  = (const float*)d_in[2];
  const float* W2  = (const float*)d_in[3];
  const float* b2  = (const float*)d_in[4];
  const float* eps = (const float*)d_in[5];
  float* out = (float*)d_out;

  char* ws = (char*)d_ws;
  __bf16* xb     = (__bf16*)(ws + WS_XB);
  __bf16* wb1    = (__bf16*)(ws + WS_WB1);
  __bf16* wb2    = (__bf16*)(ws + WS_WB2);
  float*  sq     = (float*)(ws + WS_SQ);
  int*    subi   = (int*)(ws + WS_SUBI);
  __bf16* h0     = (__bf16*)(ws + WS_H0);
  __bf16* h1     = (__bf16*)(ws + WS_H1);
  float*  h2     = (float*)(ws + WS_H2);
  float*  rowsum = (float*)(ws + WS_ROWSUM);
  float*  cand_d = (float*)(ws + WS_CANDD);
  int*    cand_i = (int*)(ws + WS_CANDI);

  k_prep_x<<<N_ROWS, 32, 0, stream>>>(x, xb, sq);
  k_conv_w<<<(2 * HID * HID) / 256, 256, 0, stream>>>(W1, W2, wb1, wb2);
  k_gram_topk<<<dim3(4, N_ROWS / 16), 32, 0, stream>>>(xb, sq, cand_d, cand_i);
  k_topk_merge<<<N_ROWS / 128, 128, 0, stream>>>(cand_d, cand_i, subi, out);
  k_pool<<<N_ROWS, 128, 0, stream>>>(x, subi, h0);
  k_gemm_lrelu<<<dim3(HID / 32, N_ROWS / 32), 32, 0, stream>>>(h0, wb1, b1, h1);
  k_gemm_f32<<<dim3(HID / 32, N_ROWS / 32), 32, 0, stream>>>(h1, wb2, b2, h2);
  k_final<<<N_ROWS, 128, 0, stream>>>(x, eps, h2, out + OUT_XOUT, rowsum);
  k_entropy<<<1, 256, 0, stream>>>(rowsum, out + OUT_ENT);
}
